// Qwen3TTSTokenizerSingleCodebookEuclideanCodebook_12524124636032
// MI455X (gfx1250) — compile-verified, matrix operand order checked by reading, not run on hardware
//
#include <hip/hip_runtime.h>

// Problem sizes (fixed by the reference)
#define BB   8
#define TT   4096
#define DD   256
#define KK   8192
#define NN   (BB * TT)            // 32768 query rows

// GEMM tiling
#define WG_THREADS   256          // 8 wave32
#define WAVES        8
#define ROWS_PER_WAVE 16
#define ROWS_PER_WG  (WAVES * ROWS_PER_WAVE)   // 128
#define CHUNK_CODES  64           // codes staged in LDS per iteration
#define NUM_CHUNKS   (KK / CHUNK_CODES)        // 128
#define D_CHUNKS     (DD / 32)                 // 8 WMMA K-steps of 32
#define CHUNK_ELEMS  (CHUNK_CODES * DD)        // 16384 ushorts = 32 KB
#define CHUNK_VEC16  (CHUNK_ELEMS / 8)         // 2048 uint4 per plane

typedef __bf16          bf16x16 __attribute__((ext_vector_type(16)));
typedef float           f32x8   __attribute__((ext_vector_type(8)));
typedef unsigned short  u16x8   __attribute__((ext_vector_type(8)));
typedef unsigned short  u16x16  __attribute__((ext_vector_type(16)));

// ---- gfx1250 async global->LDS path (guarded: falls back to plain copy) ----
#if defined(__has_builtin)
#  if __has_builtin(__builtin_amdgcn_global_load_async_to_lds_b128)
#    define HAVE_ASYNC_LDS 1
#  endif
#endif
#ifndef HAVE_ASYNC_LDS
#  define HAVE_ASYNC_LDS 0
#endif

// Parameter types per the clang diagnostic (Round 2):
//   arg0: int __attribute__((vector_size(16))) __device__ * (AS1 / global)
//   arg1: same vector type in __shared__ (AS3 / LDS)
typedef int v4i __attribute__((vector_size(16)));
typedef v4i __attribute__((address_space(1))) gv4i;   // global
typedef v4i __attribute__((address_space(3))) lv4i;   // LDS

#define TO_GLOBAL_V4I(p) ((gv4i*)(unsigned long long)(p))
#define TO_LDS_V4I(p)    ((lv4i*)(unsigned int)(unsigned long long)(p))

__device__ __forceinline__ void wait_async_all() {
#if HAVE_ASYNC_LDS
#  if __has_builtin(__builtin_amdgcn_s_wait_asynccnt)
  __builtin_amdgcn_s_wait_asynccnt(0);
#  else
  asm volatile("s_wait_asynccnt 0x0" ::: "memory");
#  endif
#endif
}

__device__ __forceinline__ unsigned short f32_to_bf16_rne(float f) {
  unsigned int u = __float_as_uint(f);
  unsigned int r = u + 0x7FFFu + ((u >> 16) & 1u);
  return (unsigned short)(r >> 16);
}
__device__ __forceinline__ float bf16_bits_to_f32(unsigned short h) {
  return __uint_as_float(((unsigned int)h) << 16);
}

// ---------------------------------------------------------------------------
// Kernel 1: codebook prep — split embed (f32) into bf16 hi/lo planes and
// compute e_sq[k] = sum_d embed[k][d]^2 in f32.
// ---------------------------------------------------------------------------
__global__ void vq_prep(const float* __restrict__ embed,
                        unsigned short* __restrict__ ehi,
                        unsigned short* __restrict__ elo,
                        float* __restrict__ esq) {
  const int row = blockIdx.x;
  const int t   = threadIdx.x;
  const size_t off = (size_t)row * DD + t;
  float v = embed[off];
  unsigned short hb = f32_to_bf16_rne(v);
  float hf = bf16_bits_to_f32(hb);
  unsigned short lb = f32_to_bf16_rne(v - hf);
  ehi[off] = hb;
  elo[off] = lb;

  __shared__ float red[256];
  red[t] = v * v;
  __syncthreads();
  #pragma unroll
  for (int s = 128; s > 0; s >>= 1) {
    if (t < s) red[t] += red[t + s];
    __syncthreads();
  }
  if (t == 0) esq[row] = red[0];
}

// ---------------------------------------------------------------------------
// Kernel 2: fused score-GEMM + argmax over the codebook.
// score(n, k) = 2 * <x_n, e_k> - ||e_k||^2    (x_n^2 is row-constant)
// Dots via V_WMMA_F32_16X16X32_BF16, 3-term hi/lo split: xh.eh + xh.el + xl.eh
// Async double-buffered LDS staging overlaps the stage of chunk kb+1 with the
// 96 WMMAs of chunk kb. Two code-tiles run on independent accumulators to
// fill the bf16-WMMA 4-slot co-exec hazard window with real work.
// ---------------------------------------------------------------------------
__device__ __forceinline__ void stage_chunk(const unsigned short* __restrict__ ehi,
                                            const unsigned short* __restrict__ elo,
                                            unsigned short* shi,
                                            unsigned short* slo,
                                            int kb, int tid) {
  const uint4* ghi = (const uint4*)(ehi + (size_t)kb * CHUNK_ELEMS);
  const uint4* glo = (const uint4*)(elo + (size_t)kb * CHUNK_ELEMS);
  uint4* dhi = (uint4*)shi;
  uint4* dlo = (uint4*)slo;
  #pragma unroll
  for (int i = 0; i < CHUNK_VEC16 / WG_THREADS; ++i) {   // 8 per plane
    int p = i * WG_THREADS + tid;
#if HAVE_ASYNC_LDS
    __builtin_amdgcn_global_load_async_to_lds_b128(
        TO_GLOBAL_V4I(ghi + p), TO_LDS_V4I(dhi + p), 0, 0);
    __builtin_amdgcn_global_load_async_to_lds_b128(
        TO_GLOBAL_V4I(glo + p), TO_LDS_V4I(dlo + p), 0, 0);
#else
    dhi[p] = ghi[p];
    dlo[p] = glo[p];
#endif
  }
}

__device__ __forceinline__ bf16x16 load_bfrag(const unsigned short* p) {
  u16x8 a = *(const u16x8*)(p);
  u16x8 b = *(const u16x8*)(p + 16);
  return __builtin_bit_cast(bf16x16,
      __builtin_shufflevector(a, b, 0,1,2,3,4,5,6,7,8,9,10,11,12,13,14,15));
}

__global__ void __launch_bounds__(WG_THREADS)
vq_argmax(const float* __restrict__ x,
          const unsigned short* __restrict__ ehi,
          const unsigned short* __restrict__ elo,
          const float* __restrict__ esq,
          int* __restrict__ idx_out) {
  __shared__ unsigned short s_ehi[2][CHUNK_ELEMS];   // 2 x 32 KB
  __shared__ unsigned short s_elo[2][CHUNK_ELEMS];   // 2 x 32 KB

  const int tid  = threadIdx.x;
  const int lane = tid & 31;
  const int wave = tid >> 5;
  const int lm   = lane & 15;     // row (A) / col (B) within the 16-tile
  const int h    = lane >> 4;     // K-half select per ISA fragment layout

  // ---- Build A fragments (hi & lo) for this wave's 16 rows, all of D ----
  // ISA 16-bit A 16x32: lane l, elems 0..7 <- K = 8h..8h+7 ; elems 8..15 <-
  // K = 16+8h .. 23+8h. Both are contiguous 32B (8 x f32) runs of the x row.
  const int m_global = blockIdx.x * ROWS_PER_WG + wave * ROWS_PER_WAVE + lm;
  const float* xr = x + (size_t)m_global * DD;

  bf16x16 a_hi[D_CHUNKS];
  bf16x16 a_lo[D_CHUNKS];
  #pragma unroll
  for (int c = 0; c < D_CHUNKS; ++c) {
    const float* p0 = xr + 32 * c + 8 * h;        // elems 0..7
    const float* p1 = xr + 32 * c + 16 + 8 * h;   // elems 8..15
    u16x16 hu, lu;
    #pragma unroll
    for (int i = 0; i < 8; ++i) {
      float f = p0[i];
      unsigned short hb = f32_to_bf16_rne(f);
      hu[i] = hb;
      lu[i] = f32_to_bf16_rne(f - bf16_bits_to_f32(hb));
    }
    #pragma unroll
    for (int i = 0; i < 8; ++i) {
      float f = p1[i];
      unsigned short hb = f32_to_bf16_rne(f);
      hu[8 + i] = hb;
      lu[8 + i] = f32_to_bf16_rne(f - bf16_bits_to_f32(hb));
    }
    a_hi[c] = __builtin_bit_cast(bf16x16, hu);
    a_lo[c] = __builtin_bit_cast(bf16x16, lu);
  }

  // Running argmax state: element j of lane l tracks row M = j + 8*(l/16).
  float best[8];
  int   bidx[8];
  #pragma unroll
  for (int j = 0; j < 8; ++j) { best[j] = -3.4e38f; bidx[j] = 0; }

  // Prologue: stage chunk 0 into buffer 0.
  stage_chunk(ehi, elo, s_ehi[0], s_elo[0], 0, tid);
  wait_async_all();
  __syncthreads();

  for (int kb = 0; kb < NUM_CHUNKS; ++kb) {
    const int cur = kb & 1;

    // Overlap: stage next chunk into the other buffer while computing.
    // (Safe: all waves finished reading buffer cur^1 at the previous barrier.)
    if (kb + 1 < NUM_CHUNKS)
      stage_chunk(ehi, elo, s_ehi[cur ^ 1], s_elo[cur ^ 1], kb + 1, tid);

    const unsigned short* sh_hi = s_ehi[cur];
    const unsigned short* sh_lo = s_elo[cur];

    // ---- 2 passes of 2 concurrent code-tiles (16 codes each) ----
    #pragma unroll
    for (int cth = 0; cth < 2; ++cth) {
      const int nl0 = (2 * cth) * 16 + lm;         // tile A: code within chunk
      const int nl1 = nl0 + 16;                    // tile B
      const unsigned short* bh0 = sh_hi + nl0 * DD + 8 * h;
      const unsigned short* bl0 = sh_lo + nl0 * DD + 8 * h;
      const unsigned short* bh1 = sh_hi + nl1 * DD + 8 * h;
      const unsigned short* bl1 = sh_lo + nl1 * DD + 8 * h;

      f32x8 acc0 = {0.f, 0.f, 0.f, 0.f, 0.f, 0.f, 0.f, 0.f};
      f32x8 acc1 = {0.f, 0.f, 0.f, 0.f, 0.f, 0.f, 0.f, 0.f};
      #pragma unroll
      for (int c = 0; c < D_CHUNKS; ++c) {
        // ISA 16-bit B 32x16: two contiguous 16B runs along K per fragment.
        bf16x16 Bh0 = load_bfrag(bh0 + 32 * c);
        bf16x16 Bl0 = load_bfrag(bl0 + 32 * c);
        bf16x16 Bh1 = load_bfrag(bh1 + 32 * c);
        bf16x16 Bl1 = load_bfrag(bl1 + 32 * c);

        // Interleave two independent accumulate chains.
        acc0 = __builtin_amdgcn_wmma_f32_16x16x32_bf16(
                   false, a_hi[c], false, Bh0, (short)0, acc0, false, false);
        acc1 = __builtin_amdgcn_wmma_f32_16x16x32_bf16(
                   false, a_hi[c], false, Bh1, (short)0, acc1, false, false);
        acc0 = __builtin_amdgcn_wmma_f32_16x16x32_bf16(
                   false, a_hi[c], false, Bl0, (short)0, acc0, false, false);
        acc1 = __builtin_amdgcn_wmma_f32_16x16x32_bf16(
                   false, a_hi[c], false, Bl1, (short)0, acc1, false, false);
        acc0 = __builtin_amdgcn_wmma_f32_16x16x32_bf16(
                   false, a_lo[c], false, Bh0, (short)0, acc0, false, false);
        acc1 = __builtin_amdgcn_wmma_f32_16x16x32_bf16(
                   false, a_lo[c], false, Bh1, (short)0, acc1, false, false);
      }

      const int g0 = kb * CHUNK_CODES + nl0;
      const int g1 = kb * CHUNK_CODES + nl1;
      const float es0 = esq[g0];
      const float es1 = esq[g1];
      #pragma unroll
      for (int j = 0; j < 8; ++j) {
        float v0 = 2.0f * acc0[j] - es0;
        if (v0 > best[j]) { best[j] = v0; bidx[j] = g0; }
        float v1 = 2.0f * acc1[j] - es1;
        if (v1 > best[j]) { best[j] = v1; bidx[j] = g1; }
      }
    }

    // Next chunk staged + everyone done with the current buffer.
    wait_async_all();
    __syncthreads();
  }

  // ---- Cross-lane argmax over the 16 columns (prefer lower index on tie,
  // matching jnp.argmax first-occurrence semantics) ----
  #pragma unroll
  for (int mask = 8; mask >= 1; mask >>= 1) {
    #pragma unroll
    for (int j = 0; j < 8; ++j) {
      float ov = __shfl_xor(best[j], mask, 16);
      int   oi = __shfl_xor(bidx[j], mask, 16);
      if (ov > best[j] || (ov == best[j] && oi < bidx[j])) {
        best[j] = ov; bidx[j] = oi;
      }
    }
  }
  if (lm == 0) {
    const int rowbase = blockIdx.x * ROWS_PER_WG + wave * ROWS_PER_WAVE + 8 * h;
    #pragma unroll
    for (int j = 0; j < 8; ++j) idx_out[rowbase + j] = bidx[j];
  }
}

// ---------------------------------------------------------------------------
// Kernel 3: decode gather — out[n,:] = embed[idx[n],:] (exact f32 values).
// ---------------------------------------------------------------------------
__global__ void vq_gather(const float* __restrict__ embed,
                          const int* __restrict__ idx,
                          float* __restrict__ out) {
  const int row = blockIdx.x;
  const int id  = idx[row];
  const float4* src = (const float4*)(embed + (size_t)id * DD);
  float4* dst = (float4*)(out + (size_t)row * DD);
  dst[threadIdx.x] = src[threadIdx.x];
}

// ---------------------------------------------------------------------------
extern "C" void kernel_launch(void* const* d_in, const int* in_sizes, int n_in,
                              void* d_out, int out_size, void* d_ws, size_t ws_size,
                              hipStream_t stream) {
  const float* x     = (const float*)d_in[0];   // [B,T,D] f32
  const float* embed = (const float*)d_in[1];   // [K,D]   f32
  float* out = (float*)d_out;                   // [B,T,D] f32

  // Workspace layout: e_hi | e_lo | e_sq | idx
  char* ws = (char*)d_ws;
  unsigned short* ehi = (unsigned short*)ws;
  unsigned short* elo = ehi + (size_t)KK * DD;
  float* esq = (float*)(ws + 2 * (size_t)KK * DD * sizeof(unsigned short));
  int*   idx = (int*)(esq + KK);

  vq_prep<<<KK, 256, 0, stream>>>(embed, ehi, elo, esq);
  vq_argmax<<<NN / ROWS_PER_WG, WG_THREADS, 0, stream>>>(x, ehi, elo, esq, idx);
  vq_gather<<<NN, DD / 4, 0, stream>>>(embed, idx, out);
}
